// net_29137058136198
// MI455X (gfx1250) — matrix operand order repaired and checked by reading, not compiled, not run on hardware
//
#include <hip/hip_runtime.h>

// ---------------- constants ----------------
#define Bz 32
#define Az 64
#define Cz 16
#define Rz 512
#define NAz (Bz*Az)      // 2048
#define NCz (Bz*Cz)      // 512
#define NRz (Bz*Rz)      // 16384
#define EAz (NAz*4)      // 8192
#define ECz (NCz*4)      // 2048
#define ERz (NRz*16)     // 262144
#define Hz 200
#define H2z 400
#define HEADSz 5
#define DHz 40

typedef __attribute__((ext_vector_type(16))) _Float16 v16h;
typedef __attribute__((ext_vector_type(8)))  float    v8f;

// ---------------- small device helpers ----------------
__device__ inline void atomicMinF(float* p, float v) {
  int* ip = (int*)p;
  int cur = __float_as_int(*p);
  while (v < __int_as_float(cur)) {
    int old = atomicCAS(ip, cur, __float_as_int(v));
    if (old == cur) break;
    cur = old;
  }
}
__device__ inline void atomicMaxF(float* p, float v) {
  int* ip = (int*)p;
  int cur = __float_as_int(*p);
  while (v > __int_as_float(cur)) {
    int old = atomicCAS(ip, cur, __float_as_int(v));
    if (old == cur) break;
    cur = old;
  }
}

// ---------------- WMMA GEMM ----------------
// C[M,N] = epi( A[M,K] @ B[K,N] ), f32 in memory, f16 WMMA with f32 accum.
// Block: 256 threads (8 wave32), tile 64x64; wave grid 2(M)x4(N), each wave
// computes a 32x16 strip = two v_wmma_f32_16x16x32_f16 per k-step (B fragment
// reused). LDS rows padded to 40 halves: 8B-aligned quad stores + conflict-free
// fragment reads. Interior tiles take a float4 fast path; edges use clamped
// loads + select (no exec-divergent per-element loads).
__global__ __launch_bounds__(256)
void k_gemm(const float* __restrict__ A, const float* __restrict__ B,
            float* __restrict__ Cout, const float* __restrict__ Cin,
            const float* __restrict__ rowscale,
            int M, int N, int K, int relu_ab, int relu_out)
{
  __shared__ _Float16 As[64][40];   // [m][k]
  __shared__ _Float16 Bs[64][40];   // [n][k] (transposed)
  const int tid  = threadIdx.x;
  const int lane = tid & 31;
  const int wave = tid >> 5;
  const int wm = wave >> 2, wn = wave & 3;     // 2 x 4 wave grid
  const int m0 = blockIdx.x * 64, n0 = blockIdx.y * 64;
  const int l15 = lane & 15, hs = lane >> 4;
  const bool mn_full = (m0 + 64 <= M) && (n0 + 64 <= N);
  const bool vecok = ((K & 3) == 0) && ((N & 3) == 0);
  v8f acc0 = {0.f,0.f,0.f,0.f,0.f,0.f,0.f,0.f};
  v8f acc1 = {0.f,0.f,0.f,0.f,0.f,0.f,0.f,0.f};
  for (int k0 = 0; k0 < K; k0 += 32) {
    if (mn_full && vecok && (k0 + 32 <= K)) {
      // ---- fast path: float4 loads, quad half stores ----
#pragma unroll
      for (int it = 0; it < 2; ++it) {
        int i = tid + it*256;                 // 512 float4 for A (64r x 8q)
        int r = i >> 3, c4 = (i & 7) << 2;
        float4 v = *(const float4*)(A + (size_t)(m0 + r)*K + k0 + c4);
        _Float16* d = &As[r][c4];
        d[0] = (_Float16)v.x; d[1] = (_Float16)v.y;
        d[2] = (_Float16)v.z; d[3] = (_Float16)v.w;
      }
#pragma unroll
      for (int it = 0; it < 2; ++it) {
        int i = tid + it*256;                 // 512 float4 for B (32k x 16q)
        int kk = i >> 4, n4 = (i & 15) << 2;
        float4 v = *(const float4*)(B + (size_t)(k0 + kk)*N + n0 + n4);
        Bs[n4+0][kk] = (_Float16)v.x; Bs[n4+1][kk] = (_Float16)v.y;
        Bs[n4+2][kk] = (_Float16)v.z; Bs[n4+3][kk] = (_Float16)v.w;
      }
    } else {
      // ---- edge path: clamped addresses + select (branch-free) ----
      for (int i = tid; i < 64*32; i += 256) {
        int r = i >> 5, c = i & 31;
        int gm = m0 + r, gk = k0 + c;
        int cm = gm < M ? gm : M - 1;
        int ck = gk < K ? gk : K - 1;
        float v = A[(size_t)cm*K + ck];
        As[r][c] = (_Float16)((gm < M && gk < K) ? v : 0.f);
      }
      for (int i = tid; i < 32*64; i += 256) {
        int kk = i >> 6, nn = i & 63;
        int gk = k0 + kk, gn = n0 + nn;
        int ckk = gk < K ? gk : K - 1;
        int cnn = gn < N ? gn : N - 1;
        float v = B[(size_t)ckk*N + cnn];
        Bs[nn][kk] = (_Float16)((gk < K && gn < N) ? v : 0.f);
      }
    }
    __syncthreads();
    // A fragment (16x32 f16): lanes 0-15 row=lane, K 0-7/16-23; lanes 16-31 +8
    v16h af0, af1, bf;
    const int ar0 = wm*32 + l15, ar1 = ar0 + 16;
    const int ab = hs*8;
#pragma unroll
    for (int h = 0; h < 16; ++h) {
      int kk = ab + ((h < 8) ? h : h + 8);
      af0[h] = As[ar0][kk];
      af1[h] = As[ar1][kk];
    }
    // B fragment (32x16): lanes 0-15 K 0-7/16-23; lanes 16-31 +8, N=lane&15
    const int bc = wn*16 + l15;
    const int bb = hs*8;
#pragma unroll
    for (int h = 0; h < 16; ++h) bf[h] = Bs[bc][bb + ((h < 8) ? h : h + 8)];
    acc0 = __builtin_amdgcn_wmma_f32_16x16x32_f16(false, af0, false, bf,
                                                  (short)0, acc0, false, false);
    acc1 = __builtin_amdgcn_wmma_f32_16x16x32_f16(false, af1, false, bf,
                                                  (short)0, acc1, false, false);
    __syncthreads();
  }
  const int col = n0 + wn*16 + l15;
  if (col < N) {
#pragma unroll
    for (int r = 0; r < 8; ++r) {
      int row0 = m0 + wm*32 + hs*8 + r;       // C layout: VGPR r -> M=r / M=8+r
      if (row0 < M) {
        float v = acc0[r];
        if (relu_ab) v = fmaxf(v, 0.f);
        if (rowscale) v *= rowscale[row0];
        if (Cin) v += Cin[(size_t)row0*N + col];
        if (relu_out) v = fmaxf(v, 0.f);
        Cout[(size_t)row0*N + col] = v;
      }
      int row1 = row0 + 16;
      if (row1 < M) {
        float v = acc1[r];
        if (relu_ab) v = fmaxf(v, 0.f);
        if (rowscale) v *= rowscale[row1];
        if (Cin) v += Cin[(size_t)row1*N + col];
        if (relu_out) v = fmaxf(v, 0.f);
        Cout[(size_t)row1*N + col] = v;
      }
    }
  }
}

// ---------------- elementwise / utility kernels ----------------
__global__ void k_fill(float* p, float v, long long n) {
  long long i = (long long)blockIdx.x*256 + threadIdx.x;
  if (i < n) p[i] = v;
}
__global__ void k_rbf(const float* __restrict__ w, float* __restrict__ out, long long n) {
  long long i = (long long)blockIdx.x*256 + threadIdx.x;
  if (i >= n) return;
  long long e = i / Hz; int j = (int)(i % Hz);
  float D = fminf(w[e], 1.0f);
  float mu = (float)j * (1.0f/199.0f);
  float z = (D - mu) * 200.0f;               // sigma = 1/200
  out[i] = __expf(-z*z);
}
__global__ __launch_bounds__(64)
void k_ln(const float* __restrict__ in, float* __restrict__ out, int W,
          int addmode, const float* __restrict__ emb, const int* __restrict__ idx)
{
  __shared__ float red[64];
  const int row = blockIdx.x, tid = threadIdx.x;
  const float* r = in + (size_t)row * W;
  float s = 0.f;
  for (int j = tid; j < W; j += 64) s += r[j];
  red[tid] = s; __syncthreads();
  for (int o = 32; o > 0; o >>= 1) { if (tid < o) red[tid] += red[tid+o]; __syncthreads(); }
  float mean = red[0] / W; __syncthreads();
  float v = 0.f;
  for (int j = tid; j < W; j += 64) { float d = r[j]-mean; v += d*d; }
  red[tid] = v; __syncthreads();
  for (int o = 32; o > 0; o >>= 1) { if (tid < o) red[tid] += red[tid+o]; __syncthreads(); }
  float rs = rsqrtf(red[0]/W + 1e-5f);
  float* w = out + (size_t)row * W;
  for (int j = tid; j < W; j += 64) {
    float val = (r[j]-mean)*rs;
    if (emb) val += emb[(size_t)idx[row]*W + j];
    if (addmode) w[j] += val; else w[j] = val;
  }
}
__global__ __launch_bounds__(256)
void k_graph_norm(const float* __restrict__ in, float* __restrict__ out, int nper, int W)
{
  __shared__ float sm[256], sr[256];
  const int b = blockIdx.x, j = threadIdx.x;
  if (j < W) {
    float s = 0.f, s2 = 0.f;
    for (int r = 0; r < nper; ++r) {
      float v = in[((size_t)b*nper + r)*W + j];
      s += v; s2 += v*v;
    }
    float mean = s / nper;
    float var = fmaxf(s2/nper - mean*mean, 0.f);
    sm[j] = mean; sr[j] = rsqrtf(var + 1e-5f);
  }
  __syncthreads();
  for (int t = threadIdx.x; t < nper*W; t += blockDim.x) {
    int r = t / W, jj = t % W;
    size_t o = ((size_t)b*nper + r)*W + jj;
    out[o] = (in[o] - sm[jj]) * sr[jj];
  }
}
__global__ __launch_bounds__(1024)
void k_bilstm(const float* __restrict__ xp, const float* __restrict__ Wh,
              float* __restrict__ out, int reverse, int out_off)
{
  __shared__ float sW[100*100];   // 40 KB
  __shared__ float h[32*100];     // 12.8 KB
  for (int i = threadIdx.x; i < 100*100; i += blockDim.x) sW[i] = Wh[i];
  for (int i = threadIdx.x; i < 32*100; i += blockDim.x) h[i] = 0.f;
  __syncthreads();
  for (int step = 0; step < 64; ++step) {
    int t = reverse ? (63 - step) : step;
    float v[4]; int c = 0;
    for (int o = threadIdx.x; o < 3200; o += blockDim.x, ++c) {
      int b = o / 100, j = o % 100;
      float s = xp[(size_t)(b*64 + t)*100 + j];
      const float* hb = &h[b*100];
      for (int k = 0; k < 100; ++k) s += hb[k] * sW[k*100 + j];
      v[c] = tanhf(s);
    }
    __syncthreads();
    c = 0;
    for (int o = threadIdx.x; o < 3200; o += blockDim.x, ++c) {
      h[o] = v[c];
      int b = o / 100, j = o % 100;
      out[(size_t)(b*64 + t)*200 + out_off + j] = v[c];
    }
    __syncthreads();
  }
}
__global__ void k_gather_emb(const int* __restrict__ idx, const float* __restrict__ emb,
                             float* __restrict__ out, long long n, int W) {
  long long i = (long long)blockIdx.x*256 + threadIdx.x;
  if (i < n) out[i] = emb[(size_t)idx[i / W]*W + (i % W)];
}
__global__ void k_deg(const int* __restrict__ dst, float* deg, long long E) {
  long long i = (long long)blockIdx.x*256 + threadIdx.x;
  if (i < E) atomicAdd(&deg[dst[i]], 1.0f);
}
__global__ void k_deg_reduce(const float* __restrict__ deg, int N, float* st) {
  int i = blockIdx.x*256 + threadIdx.x;
  if (i < N) { atomicAdd(&st[0], logf(deg[i] + 1.0f)); atomicAdd(&st[1], deg[i]); }
}
__global__ void k_scalers(const float* __restrict__ deg, const float* __restrict__ st,
                          float* amp, float* lin, int N) {
  int i = blockIdx.x*256 + threadIdx.x;
  if (i >= N) return;
  amp[i] = logf(deg[i] + 1.0f) * (float)N / st[0];
  lin[i] = deg[i] * (float)N / st[1];
}
__global__ void k_edge_msg(float* __restrict__ e, const float* __restrict__ xs,
                           const float* __restrict__ xd, const int* __restrict__ src,
                           const int* __restrict__ dst, long long n, int W) {
  long long i = (long long)blockIdx.x*256 + threadIdx.x;
  if (i >= n) return;
  long long ed = i / W; int j = (int)(i % W);
  float v = e[i] + xs[(size_t)src[ed]*W + j] + xd[(size_t)dst[ed]*W + j];
  e[i] = fmaxf(v, 0.f);
}
__global__ void k_pna_edge_agg(const float* __restrict__ m, const int* __restrict__ dst,
                               float* sum, float* sq, float* mn, float* mx,
                               long long n, int W) {
  long long i = (long long)blockIdx.x*256 + threadIdx.x;
  if (i >= n) return;
  long long e = i / W; int j = (int)(i % W);
  float v = m[i];
  size_t o = (size_t)dst[e]*W + j;
  atomicAdd(&sum[o], v); atomicAdd(&sq[o], v*v);
  atomicMinF(&mn[o], v); atomicMaxF(&mx[o], v);
}
__global__ void k_pna_node_agg(const float* __restrict__ sum, const float* __restrict__ sq,
                               const float* __restrict__ mn, const float* __restrict__ mx,
                               const float* __restrict__ deg, float* __restrict__ agg,
                               long long n, int W) {
  long long i = (long long)blockIdx.x*256 + threadIdx.x;
  if (i >= n) return;
  long long nd = i / W; int j = (int)(i % W);
  float d = deg[nd], dc = fmaxf(d, 1.f);
  float mean = sum[i] / dc;
  float st = sqrtf(fmaxf(sq[i]/dc - mean*mean, 0.f) + 1e-5f);
  float mnv = d > 0.f ? mn[i] : 0.f;
  float mxv = d > 0.f ? mx[i] : 0.f;
  size_t base = (size_t)nd * 4 * W;
  agg[base + j] = mean; agg[base + W + j] = mnv;
  agg[base + 2*W + j] = mxv; agg[base + 3*W + j] = st;
}
__global__ void k_res_relu(const float* __restrict__ x, const float* __restrict__ t,
                           float* __restrict__ out, long long n) {
  long long i = (long long)blockIdx.x*256 + threadIdx.x;
  if (i < n) out[i] = fmaxf(x[i] + t[i], 0.f);
}
__global__ void k_scatter_add(const float* __restrict__ feat, const int* __restrict__ src,
                              const int* __restrict__ dst, float* out, long long n, int W) {
  long long i = (long long)blockIdx.x*256 + threadIdx.x;
  if (i >= n) return;
  long long e = i / W; int j = (int)(i % W);
  atomicAdd(&out[(size_t)dst[e]*W + j], feat[(size_t)src[e]*W + j]);
}
__global__ void k_div_rows(float* x, const float* __restrict__ cnt, long long n, int W) {
  long long i = (long long)blockIdx.x*256 + threadIdx.x;
  if (i < n) x[i] /= fmaxf(cnt[i / W], 1.f);
}
__global__ void k_div_add(float* x, const float* __restrict__ t,
                          const float* __restrict__ deg, long long n, int W) {
  long long i = (long long)blockIdx.x*256 + threadIdx.x;
  if (i < n) x[i] += t[i] / fmaxf(deg[i / W], 1.f);
}
__global__ void k_gcn_combine(float* t, const float* __restrict__ x,
                              const float* __restrict__ deg, long long n, int W) {
  long long i = (long long)blockIdx.x*256 + threadIdx.x;
  if (i < n) t[i] = (t[i] + x[i]) / (deg[i / W] + 1.f);
}
__global__ void k_softmax_row(const float* __restrict__ in, float* __restrict__ out,
                              long long rows, int K) {
  long long i = (long long)blockIdx.x*256 + threadIdx.x;
  if (i >= rows) return;
  const float* r = in + i*K; float* w = out + i*K;
  float m = -1e30f;
  for (int k = 0; k < K; ++k) m = fmaxf(m, r[k]);
  float s = 0.f;
  for (int k = 0; k < K; ++k) { float e = __expf(r[k]-m); w[k] = e; s += e; }
  float inv = 1.f / s;
  for (int k = 0; k < K; ++k) w[k] *= inv;
}
__global__ void k_cluster_x(const float* __restrict__ s, const float* __restrict__ rx,
                            float* __restrict__ cx, int K, long long n) {
  long long i = (long long)blockIdx.x*256 + threadIdx.x;
  if (i >= n) return;
  int b = (int)(i / ((long long)K*Hz));
  long long rem = i % ((long long)K*Hz);
  int k = (int)(rem / Hz), j = (int)(rem % Hz);
  float acc = 0.f;
  for (int r = 0; r < Rz; ++r) {
    size_t nd = (size_t)b*Rz + r;
    acc += s[nd*K + k] * rx[nd*Hz + j];
  }
  cx[i] = acc;
}
__global__ void k_attn_e(const float* __restrict__ cx, const float* __restrict__ W,
                         float* __restrict__ e, int n) {
  int i = blockIdx.x*256 + threadIdx.x;
  if (i >= n) return;
  int c = i / HEADSz, h = i % HEADSz;
  float s = 0.f;
  for (int j = 0; j < Hz; ++j) s += cx[(size_t)c*Hz + j] * W[j*HEADSz + h];
  e[i] = s;
}
__global__ void k_alpha(const float* __restrict__ e, float* __restrict__ alpha) {
  int t = threadIdx.x;
  if (t >= Bz*HEADSz) return;
  int b = t / HEADSz, h = t % HEADSz;
  float m = -1e30f;
  for (int c = 0; c < Cz; ++c) m = fmaxf(m, e[(size_t)(b*Cz + c)*HEADSz + h]);
  float ex[Cz]; float s = 0.f;
  for (int c = 0; c < Cz; ++c) { ex[c] = __expf(e[(size_t)(b*Cz + c)*HEADSz + h] - m); s += ex[c]; }
  float inv = 1.f / (s + 1e-12f);
  for (int c = 0; c < Cz; ++c) alpha[(size_t)(b*Cz + c)*HEADSz + h] = ex[c] * inv;
}
__global__ void k_drug_pool(const float* __restrict__ alpha, const float* __restrict__ cx,
                            float* __restrict__ pooled, int n) {
  int i = blockIdx.x*256 + threadIdx.x;
  if (i >= n) return;
  int b = i / Hz, j = i % Hz, h = j / DHz;
  float s = 0.f;
  for (int c = 0; c < Cz; ++c) {
    size_t row = (size_t)b*Cz + c;
    s += alpha[row*HEADSz + h] * cx[row*Hz + j];
  }
  pooled[i] = s;
}
__global__ void k_att_scores(const float* __restrict__ vh, const float* __restrict__ qh,
                             float* __restrict__ att, int K, long long n) {
  long long i = (long long)blockIdx.x*256 + threadIdx.x;
  if (i >= n) return;
  long long t = i;
  int k = (int)(t % K); t /= K;
  int c = (int)(t % Cz); t /= Cz;
  int h = (int)(t % HEADSz);
  int b = (int)(t / HEADSz);
  float s = 0.f;
  const float* vp = vh + (size_t)(b*Cz + c)*Hz + h*DHz;
  const float* qp = qh + (size_t)(b*K + k)*Hz + h*DHz;
  for (int d = 0; d < DHz; ++d) s += vp[d] * qp[d];
  att[i] = s * 0.15811388300841897f;   // 1/sqrt(40)
}
__global__ __launch_bounds__(128)
void k_att_softmax(float* __restrict__ att, int K) {
  __shared__ float red[128];
  const int g = blockIdx.x;            // b*5+h
  const int n = Cz * K;
  float* p = att + (size_t)g * n;
  const int tid = threadIdx.x;
  float m = -1e30f;
  for (int i = tid; i < n; i += 128) m = fmaxf(m, p[i]);
  red[tid] = m; __syncthreads();
  for (int o = 64; o > 0; o >>= 1) { if (tid < o) red[tid] = fmaxf(red[tid], red[tid+o]); __syncthreads(); }
  m = red[0]; __syncthreads();
  float s = 0.f;
  for (int i = tid; i < n; i += 128) { float e = __expf(p[i]-m); p[i] = e; s += e; }
  red[tid] = s; __syncthreads();
  for (int o = 64; o > 0; o >>= 1) { if (tid < o) red[tid] += red[tid+o]; __syncthreads(); }
  float inv = 1.f / red[0];
  for (int i = tid; i < n; i += 128) p[i] *= inv;
}
__global__ void k_att_sums(const float* __restrict__ att, float* __restrict__ rsum,
                           float* __restrict__ csum, int K, int n) {
  int i = blockIdx.x*256 + threadIdx.x;
  if (i >= n) return;
  const int NR_ = Bz*HEADSz*Cz;
  if (i < NR_) {                         // row sums over k
    int bh = i / Cz, c = i % Cz;
    float s = 0.f;
    const float* p = att + ((size_t)bh*Cz + c)*K;
    for (int k = 0; k < K; ++k) s += p[k];
    rsum[i] = s;
  } else {                               // col sums over c
    int i2 = i - NR_;
    int bh = i2 / K, k = i2 % K;
    float s = 0.f;
    for (int c = 0; c < Cz; ++c) s += att[((size_t)bh*Cz + c)*K + k];
    csum[i2] = s;
  }
}
__global__ void k_bilinear_pool(const float* __restrict__ rsum, const float* __restrict__ csum,
                                const float* __restrict__ vh, const float* __restrict__ qh,
                                float* __restrict__ molp, float* __restrict__ protp,
                                int K, int n) {
  int i = blockIdx.x*256 + threadIdx.x;
  if (i >= n) return;
  int b = i / Hz, j = i % Hz, h = j / DHz;
  float sm = 0.f;
  for (int c = 0; c < Cz; ++c)
    sm += rsum[(size_t)(b*HEADSz + h)*Cz + c] * vh[(size_t)(b*Cz + c)*Hz + j];
  float sp = 0.f;
  for (int k = 0; k < K; ++k)
    sp += csum[(size_t)(b*HEADSz + h)*K + k] * qh[(size_t)(b*K + k)*Hz + j];
  molp[i] = sm; protp[i] = sp;
}
__global__ void k_node_update(const float* __restrict__ x, const float* __restrict__ t,
                              const float* __restrict__ g, const int* __restrict__ batch,
                              float* __restrict__ out, long long n, int W) {
  long long i = (long long)blockIdx.x*256 + threadIdx.x;
  if (i >= n) return;
  out[i] = x[i] + t[i] + g[(size_t)batch[i / W]*W + (i % W)];
}
__global__ void k_accum(float* molsum, float* protsum, const float* __restrict__ drug,
                        const float* __restrict__ molf, const float* __restrict__ protf, int n) {
  int i = blockIdx.x*256 + threadIdx.x;
  if (i >= n) return;
  molsum[i] += drug[i] + molf[i];
  protsum[i] += protf[i];
}
__global__ void k_final(const float* __restrict__ t, const float* __restrict__ reg2,
                        float* __restrict__ out) {
  int b = threadIdx.x;
  if (b >= Bz) return;
  float s = 0.f;
  for (int j = 0; j < Hz; ++j) s += t[(size_t)b*Hz + j] * reg2[j];
  out[b] = s;
}

// ---------------- host orchestration ----------------
extern "C" void kernel_launch(void* const* d_in, const int* in_sizes, int n_in,
                              void* d_out, int out_size, void* d_ws, size_t ws_size,
                              hipStream_t stream)
{
  (void)in_sizes; (void)n_in; (void)out_size; (void)ws_size;
  const int*   mol_x      = (const int*)d_in[0];
  const float* mol_x_feat = (const float*)d_in[1];
  const float* bond_x     = (const float*)d_in[2];
  const int*   aei        = (const int*)d_in[3];
  const int*   clq_idx    = (const int*)d_in[4];
  const int*   cei        = (const int*)d_in[5];
  const int*   a2ci       = (const int*)d_in[6];
  const float* res_in     = (const float*)d_in[7];
  const float* evo        = (const float*)d_in[8];
  const int*   rei        = (const int*)d_in[9];
  const float* rew        = (const float*)d_in[10];
  const int*   mol_batch  = (const int*)d_in[11];
  const int*   prot_batch = (const int*)d_in[12];
  // params flattened depth-first in dict insertion order
  auto LP = [&](int l, int j){ return (const float*)d_in[14 + l*20 + j]; };
  // 0 mol_pre1 1 mol_pre2 2 mol_post 3 prot_pre1 4 prot_pre2 5 prot_post
  // 6 clus1 7 clus2 8 Wv 9 Wq 10 Wom 11 Wop 12 a2c 13 attn 14 Wa 15 Wr
  // 16 c2a1 17 c2a2 18 c2r1 19 c2r2
  auto GP = [&](int j){ return (const float*)d_in[74 + j]; };
  // 0 atom_emb 1 clique_emb 2 atom_feat1 3 atom_feat2 4 prot_aa1 5 prot_aa2
  // 6 prot_evo1 7 prot_evo2 8 lstm_fx 9 lstm_fh 10 lstm_bx 11 lstm_bh 12 fuse
  // 13 mol_out1 14 mol_out2 15 prot_out1 16 prot_out2 17 reg1 18 reg2
  const int *ae_s = aei, *ae_d = aei + EAz;
  const int *ce_s = cei, *ce_d = cei + ECz;
  const int *ar   = a2ci, *ac  = a2ci + NAz;
  const int *re_s = rei, *re_d = rei + ERz;

  size_t off = 0;
  auto alloc = [&](size_t n){ float* p = (float*)d_ws + off; off += n; return p; };
  const size_t ERH = (size_t)ERz*Hz, NRH = (size_t)NRz*Hz, NAH = (size_t)NAz*Hz;
  float* ebuf1   = alloc(ERH);
  float* ebuf2   = alloc(ERH);
  float* residue_x = alloc(NRH);
  float* atom_x  = alloc(NAH);
  float* atom_bi = alloc(NAH);
  float* clique_x= alloc((size_t)NCz*Hz);
  float* xin     = alloc(NRH);
  float* nb1     = alloc(NRH);
  float* nb2     = alloc(NRH);
  float* sumb    = alloc(NRH);
  float* sqb     = alloc(NRH);
  float* mnb     = alloc(NRH);
  float* mxb     = alloc(NRH);
  float* aggb    = alloc(NRH*4);
  float* tpost   = alloc(NRH);
  float* hbuf    = alloc((size_t)NRz*H2z);
  float* hbuf2   = alloc((size_t)NRz*H2z);
  float* slog    = alloc((size_t)NRz*20);
  float* sbuf    = alloc((size_t)NRz*20);
  float* clus_x  = alloc((size_t)Bz*20*Hz);
  float* vbuf    = alloc((size_t)NCz*Hz);
  float* qbuf    = alloc((size_t)Bz*20*Hz);
  float* cagg    = alloc((size_t)NCz*Hz);
  float* ctmp    = alloc((size_t)NCz*Hz);
  float* e_attn  = alloc(NCz*HEADSz);
  float* alpha   = alloc(NCz*HEADSz);
  float* att     = alloc((size_t)Bz*HEADSz*Cz*20);
  float* rsum    = alloc(Bz*HEADSz*Cz);
  float* csum    = alloc(Bz*HEADSz*20);
  float* molp    = alloc(Bz*Hz);
  float* protp   = alloc(Bz*Hz);
  float* mol_f   = alloc(Bz*Hz);
  float* prot_f  = alloc(Bz*Hz);
  float* poolb   = alloc(Bz*Hz);
  float* drug_x  = alloc(Bz*Hz);
  float* ga      = alloc(Bz*Hz);
  float* gr      = alloc(Bz*Hz);
  float* g1      = alloc(Bz*H2z);
  float* molsum  = alloc(Bz*Hz);
  float* protsum = alloc(Bz*Hz);
  float* t2a     = alloc(Bz*H2z);
  float* t2b     = alloc(Bz*H2z);
  float* xf      = alloc((size_t)NAz*100);
  float* xb      = alloc((size_t)NAz*100);
  float* atom_deg= alloc(NAz);
  float* res_deg = alloc(NRz);
  float* cdeg    = alloc(NCz);
  float* a2cc    = alloc(NCz);
  float* ampA    = alloc(NAz);
  float* linA    = alloc(NAz);
  float* ampR    = alloc(NRz);
  float* linR    = alloc(NRz);
  float* stats   = alloc(4);

  auto NB = [](long long n){ return dim3((unsigned)((n + 255) / 256)); };
  auto gemm = [&](const float* A, const float* Bm, float* Co, const float* Ci,
                  const float* rs, int M, int N, int K, int rab, int rout){
    dim3 grid((M + 63) / 64, (N + 63) / 64);
    k_gemm<<<grid, 256, 0, stream>>>(A, Bm, Co, Ci, rs, M, N, K, rab, rout);
  };

  // ===== graph-static degrees + PNA scalers =====
  k_fill<<<NB(NAz),256,0,stream>>>(atom_deg, 0.f, NAz);
  k_deg<<<NB(EAz),256,0,stream>>>(ae_d, atom_deg, EAz);
  k_fill<<<NB(NRz),256,0,stream>>>(res_deg, 0.f, NRz);
  k_deg<<<NB(ERz),256,0,stream>>>(re_d, res_deg, ERz);
  k_fill<<<NB(NCz),256,0,stream>>>(cdeg, 0.f, NCz);
  k_deg<<<NB(ECz),256,0,stream>>>(ce_d, cdeg, ECz);
  k_fill<<<NB(NCz),256,0,stream>>>(a2cc, 0.f, NCz);
  k_deg<<<NB(NAz),256,0,stream>>>(ac, a2cc, NAz);
  k_fill<<<NB(4),256,0,stream>>>(stats, 0.f, 4);
  k_deg_reduce<<<NB(NAz),256,0,stream>>>(atom_deg, NAz, stats);
  k_deg_reduce<<<NB(NRz),256,0,stream>>>(res_deg, NRz, stats + 2);
  k_scalers<<<NB(NAz),256,0,stream>>>(atom_deg, stats, ampA, linA, NAz);
  k_scalers<<<NB(NRz),256,0,stream>>>(res_deg, stats + 2, ampR, linR, NRz);

  // ===== residue_x = ln(mlp(res_in)) + ln(mlp(evo)) =====
  gemm(res_in, GP(4), hbuf, nullptr, nullptr, NRz, H2z, 33, 0, 1);
  gemm(hbuf, GP(5), tpost, nullptr, nullptr, NRz, Hz, H2z, 0, 0);
  k_ln<<<NRz,64,0,stream>>>(tpost, residue_x, Hz, 0, nullptr, nullptr);
  gemm(evo, GP(6), hbuf, nullptr, nullptr, NRz, H2z, 1280, 0, 1);
  gemm(hbuf, GP(7), tpost, nullptr, nullptr, NRz, Hz, H2z, 0, 0);
  k_ln<<<NRz,64,0,stream>>>(tpost, residue_x, Hz, 1, nullptr, nullptr);

  // ===== atom_x = emb[mol_x] + ln(mlp(feat)) =====
  gemm(mol_x_feat, GP(2), hbuf, nullptr, nullptr, NAz, H2z, 43, 0, 1);
  gemm(hbuf, GP(3), tpost, nullptr, nullptr, NAz, Hz, H2z, 0, 0);
  k_ln<<<NAz,64,0,stream>>>(tpost, atom_x, Hz, 0, GP(0), mol_x);

  // ===== BiLSTM over atoms =====
  gemm(atom_x, GP(8),  xf, nullptr, nullptr, NAz, 100, Hz, 0, 0);
  gemm(atom_x, GP(10), xb, nullptr, nullptr, NAz, 100, Hz, 0, 0);
  k_bilstm<<<1,1024,0,stream>>>(xf, GP(9),  atom_bi, 0, 0);
  k_bilstm<<<1,1024,0,stream>>>(xb, GP(11), atom_bi, 1, 100);

  // ===== clique init, feat accumulators =====
  k_gather_emb<<<NB((long long)NCz*Hz),256,0,stream>>>(clq_idx, GP(1), clique_x, (long long)NCz*Hz, Hz);
  k_fill<<<NB(Bz*Hz),256,0,stream>>>(molsum, 0.f, Bz*Hz);
  k_fill<<<NB(Bz*Hz),256,0,stream>>>(protsum, 0.f, Bz*Hz);

  const int KSh[3] = {5, 10, 20};
  for (int i = 0; i < 3; ++i) {
    const int Ki = KSh[i];
    // ===== atom PNA + fuse =====
    gemm(atom_x, LP(i,0),          nb1, nullptr, nullptr, NAz, Hz, Hz, 0, 0);
    gemm(atom_x, LP(i,0)+Hz*Hz,    nb2, nullptr, nullptr, NAz, Hz, Hz, 0, 0);
    gemm(bond_x, LP(i,0)+2*Hz*Hz,  ebuf1, nullptr, nullptr, EAz, Hz, Hz, 0, 0);
    k_edge_msg<<<NB((long long)EAz*Hz),256,0,stream>>>(ebuf1, nb1, nb2, ae_s, ae_d, (long long)EAz*Hz, Hz);
    gemm(ebuf1, LP(i,1), ebuf2, nullptr, nullptr, EAz, Hz, Hz, 0, 0);
    k_fill<<<NB(NAH),256,0,stream>>>(sumb, 0.f, NAH);
    k_fill<<<NB(NAH),256,0,stream>>>(sqb, 0.f, NAH);
    k_fill<<<NB(NAH),256,0,stream>>>(mnb,  3.0e38f, NAH);
    k_fill<<<NB(NAH),256,0,stream>>>(mxb, -3.0e38f, NAH);
    k_pna_edge_agg<<<NB((long long)EAz*Hz),256,0,stream>>>(ebuf2, ae_d, sumb, sqb, mnb, mxb, (long long)EAz*Hz, Hz);
    k_pna_node_agg<<<NB(NAH),256,0,stream>>>(sumb, sqb, mnb, mxb, atom_deg, aggb, NAH, Hz);
    gemm(aggb, LP(i,2),          tpost, nullptr, nullptr, NAz, Hz, 4*Hz, 0, 0);
    gemm(aggb, LP(i,2)+4*Hz*Hz,  tpost, tpost, ampA, NAz, Hz, 4*Hz, 0, 0);
    gemm(aggb, LP(i,2)+8*Hz*Hz,  tpost, tpost, linA, NAz, Hz, 4*Hz, 0, 0);
    k_res_relu<<<NB(NAH),256,0,stream>>>(atom_x, tpost, atom_x, NAH);
    gemm(atom_x,  GP(12),        tpost, nullptr, nullptr, NAz, Hz, Hz, 0, 0);
    gemm(atom_bi, GP(12)+Hz*Hz,  atom_x, tpost, nullptr, NAz, Hz, Hz, 0, 1);

    // ===== residue PNA =====
    gemm(residue_x, LP(i,3),        nb1, nullptr, nullptr, NRz, Hz, Hz, 0, 0);
    gemm(residue_x, LP(i,3)+Hz*Hz,  nb2, nullptr, nullptr, NRz, Hz, Hz, 0, 0);
    k_rbf<<<NB((long long)ERH),256,0,stream>>>(rew, ebuf2, (long long)ERH);
    gemm(ebuf2, LP(i,3)+2*Hz*Hz, ebuf1, nullptr, nullptr, ERz, Hz, Hz, 0, 0);
    k_edge_msg<<<NB((long long)ERH),256,0,stream>>>(ebuf1, nb1, nb2, re_s, re_d, (long long)ERH, Hz);
    gemm(ebuf1, LP(i,4), ebuf2, nullptr, nullptr, ERz, Hz, Hz, 0, 0);
    k_fill<<<NB(NRH),256,0,stream>>>(sumb, 0.f, NRH);
    k_fill<<<NB(NRH),256,0,stream>>>(sqb, 0.f, NRH);
    k_fill<<<NB(NRH),256,0,stream>>>(mnb,  3.0e38f, NRH);
    k_fill<<<NB(NRH),256,0,stream>>>(mxb, -3.0e38f, NRH);
    k_pna_edge_agg<<<NB((long long)ERH),256,0,stream>>>(ebuf2, re_d, sumb, sqb, mnb, mxb, (long long)ERH, Hz);
    k_pna_node_agg<<<NB(NRH),256,0,stream>>>(sumb, sqb, mnb, mxb, res_deg, aggb, NRH, Hz);
    gemm(aggb, LP(i,5),          tpost, nullptr, nullptr, NRz, Hz, 4*Hz, 0, 0);
    gemm(aggb, LP(i,5)+4*Hz*Hz,  tpost, tpost, ampR, NRz, Hz, 4*Hz, 0, 0);
    gemm(aggb, LP(i,5)+8*Hz*Hz,  tpost, tpost, linR, NRz, Hz, 4*Hz, 0, 0);
    k_res_relu<<<NB(NRH),256,0,stream>>>(residue_x, tpost, residue_x, NRH);

    // ===== motif pooling -> drug vector =====
    k_fill<<<NB((long long)NCz*Hz),256,0,stream>>>(cagg, 0.f, (long long)NCz*Hz);
    k_scatter_add<<<NB(NAH),256,0,stream>>>(atom_x, ar, ac, cagg, NAH, Hz);
    k_div_rows<<<NB((long long)NCz*Hz),256,0,stream>>>(cagg, a2cc, (long long)NCz*Hz, Hz);
    gemm(cagg, LP(i,12), clique_x, clique_x, nullptr, NCz, Hz, Hz, 1, 0); // += relu(AB)
    k_fill<<<NB((long long)NCz*Hz),256,0,stream>>>(ctmp, 0.f, (long long)NCz*Hz);
    k_scatter_add<<<NB((long long)ECz*Hz),256,0,stream>>>(clique_x, ce_s, ce_d, ctmp, (long long)ECz*Hz, Hz);
    k_div_add<<<NB((long long)NCz*Hz),256,0,stream>>>(clique_x, ctmp, cdeg, (long long)NCz*Hz, Hz);
    k_attn_e<<<NB(NCz*HEADSz),256,0,stream>>>(clique_x, LP(i,13), e_attn, NCz*HEADSz);
    k_alpha<<<1,256,0,stream>>>(e_attn, alpha);
    k_drug_pool<<<NB(Bz*Hz),256,0,stream>>>(alpha, clique_x, poolb, Bz*Hz);
    k_ln<<<Bz,64,0,stream>>>(poolb, drug_x, Hz, 0, nullptr, nullptr);

    // ===== cluster assignment (2-layer GCN) + pooling =====
    k_ln<<<NRz,64,0,stream>>>(residue_x, xin, Hz, 0, nullptr, nullptr);
    k_fill<<<NB(NRH),256,0,stream>>>(nb1, 0.f, NRH);
    k_scatter_add<<<NB((long long)ERH),256,0,stream>>>(xin, re_s, re_d, nb1, (long long)ERH, Hz);
    k_gcn_combine<<<NB(NRH),256,0,stream>>>(nb1, xin, res_deg, NRH, Hz);
    gemm(nb1, LP(i,6), hbuf, nullptr, nullptr, NRz, H2z, Hz, 0, 1);
    k_fill<<<NB((long long)NRz*H2z),256,0,stream>>>(hbuf2, 0.f, (long long)NRz*H2z);
    k_scatter_add<<<NB((long long)ERz*H2z),256,0,stream>>>(hbuf, re_s, re_d, hbuf2, (long long)ERz*H2z, H2z);
    k_gcn_combine<<<NB((long long)NRz*H2z),256,0,stream>>>(hbuf2, hbuf, res_deg, (long long)NRz*H2z, H2z);
    gemm(hbuf2, LP(i,7), slog, nullptr, nullptr, NRz, Ki, H2z, 0, 0);
    k_softmax_row<<<NB(NRz),256,0,stream>>>(slog, sbuf, NRz, Ki);
    k_cluster_x<<<NB((long long)Bz*Ki*Hz),256,0,stream>>>(sbuf, residue_x, clus_x, Ki, (long long)Bz*Ki*Hz);

    // ===== bilinear cross-modal fusion =====
    gemm(clique_x, LP(i,8), vbuf, nullptr, nullptr, NCz, Hz, Hz, 0, 0);
    gemm(clus_x,   LP(i,9), qbuf, nullptr, nullptr, Bz*Ki, Hz, Hz, 0, 0);
    k_att_scores<<<NB((long long)Bz*HEADSz*Cz*Ki),256,0,stream>>>(vbuf, qbuf, att, Ki, (long long)Bz*HEADSz*Cz*Ki);
    k_att_softmax<<<Bz*HEADSz,128,0,stream>>>(att, Ki);
    k_att_sums<<<NB(Bz*HEADSz*(Cz+Ki)),256,0,stream>>>(att, rsum, csum, Ki, Bz*HEADSz*(Cz+Ki));
    k_bilinear_pool<<<NB(Bz*Hz),256,0,stream>>>(rsum, csum, vbuf, qbuf, molp, protp, Ki, Bz*Hz);
    gemm(molp,  LP(i,10), mol_f,  nullptr, nullptr, Bz, Hz, Hz, 0, 0);
    gemm(protp, LP(i,11), prot_f, nullptr, nullptr, Bz, Hz, Hz, 0, 0);

    // ===== broadcast context back + GraphNorm =====
    gemm(mol_f, LP(i,16), g1, nullptr, nullptr, Bz, H2z, Hz, 0, 1);
    gemm(g1,    LP(i,17), ga, nullptr, nullptr, Bz, Hz, H2z, 0, 0);
    gemm(atom_x, LP(i,14), tpost, nullptr, nullptr, NAz, Hz, Hz, 0, 1);
    k_node_update<<<NB(NAH),256,0,stream>>>(atom_x, tpost, ga, mol_batch, nb1, NAH, Hz);
    k_graph_norm<<<Bz,256,0,stream>>>(nb1, atom_x, Az, Hz);
    gemm(prot_f, LP(i,18), g1, nullptr, nullptr, Bz, H2z, Hz, 0, 1);
    gemm(g1,     LP(i,19), gr, nullptr, nullptr, Bz, Hz, H2z, 0, 0);
    gemm(residue_x, LP(i,15), tpost, nullptr, nullptr, NRz, Hz, Hz, 0, 1);
    k_node_update<<<NB(NRH),256,0,stream>>>(residue_x, tpost, gr, prot_batch, nb1, NRH, Hz);
    k_graph_norm<<<Bz,256,0,stream>>>(nb1, residue_x, Rz, Hz);

    k_accum<<<NB(Bz*Hz),256,0,stream>>>(molsum, protsum, drug_x, mol_f, prot_f, Bz*Hz);
  }

  // ===== output head =====
  gemm(molsum, GP(13), t2a, nullptr, nullptr, Bz, H2z, Hz, 0, 1);
  gemm(t2a,    GP(14), t2b, nullptr, nullptr, Bz, Hz, H2z, 0, 0);
  k_ln<<<Bz,64,0,stream>>>(t2b, molp, Hz, 0, nullptr, nullptr);
  gemm(protsum, GP(15), t2a, nullptr, nullptr, Bz, H2z, Hz, 0, 1);
  gemm(t2a,     GP(16), t2b, nullptr, nullptr, Bz, Hz, H2z, 0, 0);
  k_ln<<<Bz,64,0,stream>>>(t2b, protp, Hz, 0, nullptr, nullptr);
  gemm(molp,  GP(17),        t2a, nullptr, nullptr, Bz, Hz, Hz, 0, 0);
  gemm(protp, GP(17)+Hz*Hz,  t2a, t2a, nullptr, Bz, Hz, Hz, 0, 1);
  k_final<<<1,32,0,stream>>>(t2a, GP(18), (float*)d_out);
}
